// GATv2Layer_773094113320
// MI455X (gfx1250) — compile-verified, matrix operand order checked by reading, not run on hardware
//
#include <hip/hip_runtime.h>
#include <math.h>

#define NN   50000
#define FF   128
#define HH   8
#define DD   16
#define HDIM 128
#define EE   800000
#define GG   512

typedef __attribute__((ext_vector_type(16))) _Float16 v16h;
typedef __attribute__((ext_vector_type(8)))  _Float16 v8h;
typedef __attribute__((ext_vector_type(8)))  float    v8f;

// ---------------------------------------------------------------------------
// Per-lane f16 fragment loader for v_wmma_f32_16x16x32_f16.
// ISA layout (16-bit A 16x32, B 32x16 mirrored):
//   lanes 0-15 : K = k0+[0..7]  in halves 0-7,  K = k0+[16..23] in halves 8-15
//   lanes 16-31: K = k0+[8..15] in halves 0-7,  K = k0+[24..31] in halves 8-15
// Two b128 loads per fragment, no conversion in the hot loop.
// ---------------------------------------------------------------------------
__device__ __forceinline__ v16h load_row16h(const _Float16* __restrict__ row,
                                            int k0, int hi) {
    const int base = k0 + (hi ? 8 : 0);
    const v8h lo = *(const v8h*)(row + base);
    const v8h up = *(const v8h*)(row + base + 16);
    v16h r;
#pragma unroll
    for (int t = 0; t < 8; ++t) { r[t] = lo[t]; r[8 + t] = up[t]; }
    return r;
}

__device__ __forceinline__ float atomicMaxFloat(float* addr, float value) {
    if (value >= 0.0f)
        return __int_as_float(atomicMax((int*)addr, __float_as_int(value)));
    else
        return __uint_as_float(atomicMin((unsigned int*)addr, __float_as_uint(value)));
}

// ---------------------------------------------------------------------------
// K0a: generic f32 -> f16 conversion (one shot per operand, removes all
// per-tile v_cvt work from the WMMA inner loops).
// ---------------------------------------------------------------------------
__global__ void cvt_f32_to_f16_kernel(const float* __restrict__ src,
                                      _Float16* __restrict__ dst, int n) {
    const int i = blockIdx.x * blockDim.x + threadIdx.x;
    if (i < n) dst[i] = (_Float16)src[i];
}

// ---------------------------------------------------------------------------
// K0b: initialize accumulators (harness poisons buffers; re-zero every call
// so graph replays are deterministic).
// d_out layout: [0, NN*HDIM) = node out (used first as aggregation accum),
//               [NN*HDIM, +GG*256) = readout, per graph: 128 sum | 128 max.
// ---------------------------------------------------------------------------
__global__ void gat_init_kernel(float* __restrict__ d_out, float* __restrict__ denom) {
    const long long i = (long long)blockIdx.x * blockDim.x + threadIdx.x;
    if (i < (long long)NN * HDIM) d_out[i] = 0.0f;
    if (i < (long long)GG * 256) {
        const int j = (int)(i & 255);
        d_out[(long long)NN * HDIM + i] = (j < 128) ? 0.0f : -3.402823466e38f;
    }
    if (i < (long long)NN * HH) denom[i] = 0.0f;
}

// ---------------------------------------------------------------------------
// K1: projections  src_p = x @ W_src^T,  dst_p = x @ W_dst^T   (WMMA, f16 in)
// One wave per 16x16 output tile; tiles = (NN/16) * (HDIM/16) * 2 = 50000.
// ---------------------------------------------------------------------------
__global__ void gat_proj_kernel(const _Float16* __restrict__ xh,
                                const _Float16* __restrict__ Wsrch,
                                const _Float16* __restrict__ Wdsth,
                                float* __restrict__ src_p,
                                float* __restrict__ dst_p) {
    const int wave = (int)((blockIdx.x * blockDim.x + threadIdx.x) >> 5);
    const int lane = threadIdx.x & 31;
    const int mTiles = NN / 16;           // 3125
    const int nTiles = HDIM / 16;         // 8
    const int which  = wave / (mTiles * nTiles);
    if (which >= 2) return;
    const int rem = wave % (mTiles * nTiles);
    const int mT = rem / nTiles, nT = rem % nTiles;

    const _Float16* __restrict__ W    = which ? Wdsth : Wsrch;
    float*          __restrict__ outp = which ? dst_p : src_p;

    const int l16 = lane & 15;
    const int hi  = lane >> 4;
    const _Float16* xrow = xh + (size_t)(mT * 16 + l16) * FF;
    const _Float16* wrow = W  + (size_t)(nT * 16 + l16) * FF;

    v8f c = {};
#pragma unroll
    for (int k0 = 0; k0 < FF; k0 += 32) {
        v16h a = load_row16h(xrow, k0, hi);
        v16h b = load_row16h(wrow, k0, hi);
        c = __builtin_amdgcn_wmma_f32_16x16x32_f16(false, a, false, b,
                                                   (short)0, c, false, false);
    }
    const int rowBase = mT * 16 + hi * 8;
    const int col     = nT * 16 + l16;
#pragma unroll
    for (int r = 0; r < 8; ++r)
        outp[(size_t)(rowBase + r) * HDIM + col] = c[r];
}

// ---------------------------------------------------------------------------
// K2: edge logits -> exp, accumulate softmax denominator per (dst, head).
// One wave per edge; lane l handles elements [4l,4l+4) => head h = l/4.
// ---------------------------------------------------------------------------
__global__ void gat_edge_logits_kernel(const float* __restrict__ src_p,
                                       const float* __restrict__ dst_p,
                                       const int* __restrict__ src,
                                       const int* __restrict__ dst,
                                       const float* __restrict__ attn,
                                       float* __restrict__ expv,
                                       float* __restrict__ denom) {
    const int e    = (int)((blockIdx.x * blockDim.x + threadIdx.x) >> 5);
    const int lane = threadIdx.x & 31;
    if (e >= EE) return;
    const int s = src[e], d = dst[e];

    const float4 sv = *(const float4*)(src_p + (size_t)s * HDIM + lane * 4);
    const float4 dv = *(const float4*)(dst_p + (size_t)d * HDIM + lane * 4);
    const float4 av = *(const float4*)(attn + lane * 4);

    float ex0 = sv.x + dv.x, ex1 = sv.y + dv.y, ex2 = sv.z + dv.z, ex3 = sv.w + dv.w;
    ex0 = ex0 > 0.0f ? ex0 : 0.2f * ex0;
    ex1 = ex1 > 0.0f ? ex1 : 0.2f * ex1;
    ex2 = ex2 > 0.0f ? ex2 : 0.2f * ex2;
    ex3 = ex3 > 0.0f ? ex3 : 0.2f * ex3;
    float partial = ex0 * av.x + ex1 * av.y + ex2 * av.z + ex3 * av.w;
    partial += __shfl_xor(partial, 1, 32);   // reduce over the 4 lanes of a head
    partial += __shfl_xor(partial, 2, 32);
    if ((lane & 3) == 0) {
        const int h = lane >> 2;
        const float exl = expf(partial);     // global-max shift omitted: alpha invariant
        expv[(size_t)e * HH + h] = exl;
        atomicAdd(&denom[(size_t)d * HH + h], exl);
    }
}

// ---------------------------------------------------------------------------
// K3: alpha = exp/denom; scatter-add alpha * src_p[src] into accum (= d_out).
// ---------------------------------------------------------------------------
__global__ void gat_edge_aggregate_kernel(const float* __restrict__ src_p,
                                          const int* __restrict__ src,
                                          const int* __restrict__ dst,
                                          const float* __restrict__ expv,
                                          const float* __restrict__ denom,
                                          float* __restrict__ accum) {
    const int e    = (int)((blockIdx.x * blockDim.x + threadIdx.x) >> 5);
    const int lane = threadIdx.x & 31;
    if (e >= EE) return;
    const int s = src[e], d = dst[e];
    const int h = lane >> 2;
    const float alpha = expv[(size_t)e * HH + h] /
                        (denom[(size_t)d * HH + h] + 1e-16f);
    const float4 sv = *(const float4*)(src_p + (size_t)s * HDIM + lane * 4);
    float* base = accum + (size_t)d * HDIM + lane * 4;
    atomicAdd(base + 0, sv.x * alpha);
    atomicAdd(base + 1, sv.y * alpha);
    atomicAdd(base + 2, sv.z * alpha);
    atomicAdd(base + 3, sv.w * alpha);
}

// ---------------------------------------------------------------------------
// K4: out = PReLU(accum + x + bias); sigmoid gate; g_max pooling; and fused
// f16 copy of `out` for the downstream WMMA score GEMM.
// One wave per node; lane covers 4 features.
// ---------------------------------------------------------------------------
__global__ void gat_node_finalize_kernel(float* __restrict__ out,       // d_out[:NN*HDIM], in place
                                         _Float16* __restrict__ out_h,  // f16 mirror for K5
                                         const float* __restrict__ x,
                                         const float* __restrict__ bias,
                                         const float* __restrict__ prelu_alpha,
                                         const float* __restrict__ ww,
                                         const float* __restrict__ bw,
                                         const int* __restrict__ batch,
                                         float* __restrict__ weighted,
                                         float* __restrict__ readout) { // d_out + NN*HDIM
    const int n    = (int)((blockIdx.x * blockDim.x + threadIdx.x) >> 5);
    const int lane = threadIdx.x & 31;
    if (n >= NN) return;
    const float a = prelu_alpha[0];

    float*       orow = out + (size_t)n * HDIM + lane * 4;
    const float4 acc  = *(const float4*)orow;
    const float4 xv   = *(const float4*)(x + (size_t)n * HDIM + lane * 4);
    const float4 bv   = *(const float4*)(bias + lane * 4);

    float o0 = acc.x + xv.x + bv.x, o1 = acc.y + xv.y + bv.y;
    float o2 = acc.z + xv.z + bv.z, o3 = acc.w + xv.w + bv.w;
    o0 = o0 > 0.0f ? o0 : a * o0;
    o1 = o1 > 0.0f ? o1 : a * o1;
    o2 = o2 > 0.0f ? o2 : a * o2;
    o3 = o3 > 0.0f ? o3 : a * o3;
    orow[0] = o0; orow[1] = o1; orow[2] = o2; orow[3] = o3;

    _Float16* oh = out_h + (size_t)n * HDIM + lane * 4;
    oh[0] = (_Float16)o0; oh[1] = (_Float16)o1;
    oh[2] = (_Float16)o2; oh[3] = (_Float16)o3;

    // gate = sigmoid(out . w_weighting + b)
    const float4 wv = *(const float4*)(ww + lane * 4);
    float p = o0 * wv.x + o1 * wv.y + o2 * wv.z + o3 * wv.w;
#pragma unroll
    for (int m = 1; m < 32; m <<= 1) p += __shfl_xor(p, m, 32);
    if (lane == 0)
        weighted[n] = 1.0f / (1.0f + expf(-(p + bw[0])));

    // per-graph max pool
    const int g = batch[n];
    float* gm = readout + (size_t)g * 256 + 128 + lane * 4;
    atomicMaxFloat(gm + 0, o0);
    atomicMaxFloat(gm + 1, o1);
    atomicMaxFloat(gm + 2, o2);
    atomicMaxFloat(gm + 3, o3);
}

// ---------------------------------------------------------------------------
// K5: score = out @ w_score^T + b_score (WMMA, f16 in), fused gated
// segment-sum:  g_sum[batch[n]] += weighted[n] * score[n].
// One wave per 16x16 tile; tiles = (NN/16)*(HDIM/16) = 25000.
// ---------------------------------------------------------------------------
__global__ void gat_score_kernel(const _Float16* __restrict__ out_h,
                                 const _Float16* __restrict__ wscoreh,
                                 const float* __restrict__ bscore,
                                 const float* __restrict__ weighted,
                                 const int* __restrict__ batch,
                                 float* __restrict__ readout) { // d_out + NN*HDIM
    const int wave = (int)((blockIdx.x * blockDim.x + threadIdx.x) >> 5);
    const int lane = threadIdx.x & 31;
    const int nTiles = HDIM / 16;   // 8
    if (wave >= (NN / 16) * nTiles) return;
    const int mT = wave / nTiles, nT = wave % nTiles;

    const int l16 = lane & 15;
    const int hi  = lane >> 4;
    const _Float16* xrow = out_h   + (size_t)(mT * 16 + l16) * HDIM;
    const _Float16* wrow = wscoreh + (size_t)(nT * 16 + l16) * HDIM;

    v8f c = {};
#pragma unroll
    for (int k0 = 0; k0 < HDIM; k0 += 32) {
        v16h va = load_row16h(xrow, k0, hi);
        v16h vb = load_row16h(wrow, k0, hi);
        c = __builtin_amdgcn_wmma_f32_16x16x32_f16(false, va, false, vb,
                                                   (short)0, c, false, false);
    }
    const int rowBase = mT * 16 + hi * 8;
    const int col     = nT * 16 + l16;
    const float bcol  = bscore[col];
#pragma unroll
    for (int r = 0; r < 8; ++r) {
        const int row = rowBase + r;
        const float val = (c[r] + bcol) * weighted[row];
        atomicAdd(&readout[(size_t)batch[row] * 256 + col], val);
    }
}

// ---------------------------------------------------------------------------
extern "C" void kernel_launch(void* const* d_in, const int* in_sizes, int n_in,
                              void* d_out, int out_size, void* d_ws, size_t ws_size,
                              hipStream_t stream) {
    const float* x        = (const float*)d_in[0];
    const int*   eidx     = (const int*)d_in[1];   // [2,E] flat: src then dst
    const int*   batch    = (const int*)d_in[2];
    const float* W_src    = (const float*)d_in[3];
    const float* W_dst    = (const float*)d_in[4];
    const float* attn     = (const float*)d_in[5]; // (1,H,D) -> 128 floats
    const float* bias     = (const float*)d_in[6];
    const float* prelu_a  = (const float*)d_in[7];
    const float* w_weight = (const float*)d_in[8];
    const float* b_weight = (const float*)d_in[9];
    const float* w_score  = (const float*)d_in[10];
    const float* b_score  = (const float*)d_in[11];

    const int* src = eidx;
    const int* dst = eidx + EE;

    float* outbuf  = (float*)d_out;                       // [NN*HDIM] node out / accum
    float* readout = outbuf + (size_t)NN * HDIM;          // [GG*256] sum|max

    // workspace carve-up
    char* p = (char*)d_ws;
    float* src_p    = (float*)p;            p += (size_t)NN * HDIM * 4;
    float* dst_p    = (float*)p;            p += (size_t)NN * HDIM * 4;
    float* expv     = (float*)p;            p += (size_t)EE * HH * 4;
    float* denom    = (float*)p;            p += (size_t)NN * HH * 4;
    float* weighted = (float*)p;            p += (size_t)NN * 4;
    _Float16* xh       = (_Float16*)p;      p += (size_t)NN * FF * 2;
    _Float16* out_h    = (_Float16*)p;      p += (size_t)NN * HDIM * 2;
    _Float16* Wsrc_h   = (_Float16*)p;      p += (size_t)HDIM * FF * 2;
    _Float16* Wdst_h   = (_Float16*)p;      p += (size_t)HDIM * FF * 2;
    _Float16* Wscore_h = (_Float16*)p;      p += (size_t)HDIM * HDIM * 2;

    const dim3 blk(256);

    // K0a: one-shot f32 -> f16 operand conversion
    cvt_f32_to_f16_kernel<<<dim3((NN * FF + 255) / 256), blk, 0, stream>>>(x, xh, NN * FF);
    cvt_f32_to_f16_kernel<<<dim3((HDIM * FF + 255) / 256), blk, 0, stream>>>(W_src, Wsrc_h, HDIM * FF);
    cvt_f32_to_f16_kernel<<<dim3((HDIM * FF + 255) / 256), blk, 0, stream>>>(W_dst, Wdst_h, HDIM * FF);
    cvt_f32_to_f16_kernel<<<dim3((HDIM * HDIM + 255) / 256), blk, 0, stream>>>(w_score, Wscore_h, HDIM * HDIM);

    // K0b: init accumulators (covers NN*HDIM, GG*256, NN*HH ranges)
    gat_init_kernel<<<dim3((NN * HDIM + 255) / 256), blk, 0, stream>>>(outbuf, denom);

    // K1: projections (50000 tiles, 8 waves/block)
    gat_proj_kernel<<<dim3(50000 / 8), blk, 0, stream>>>(xh, Wsrc_h, Wdst_h, src_p, dst_p);

    // K2: edge logits + denom (1 wave/edge)
    gat_edge_logits_kernel<<<dim3(EE / 8), blk, 0, stream>>>(src_p, dst_p, src, dst,
                                                             attn, expv, denom);

    // K3: weighted aggregation into outbuf
    gat_edge_aggregate_kernel<<<dim3(EE / 8), blk, 0, stream>>>(src_p, src, dst,
                                                                expv, denom, outbuf);

    // K4: residual + PReLU + gate + max pool + f16 mirror (1 wave/node)
    gat_node_finalize_kernel<<<dim3((NN + 7) / 8), blk, 0, stream>>>(
        outbuf, out_h, x, bias, prelu_a, w_weight, b_weight, batch, weighted, readout);

    // K5: score GEMM + gated segment sum (25000 tiles)
    gat_score_kernel<<<dim3(25000 / 8), blk, 0, stream>>>(out_h, Wscore_h, b_score,
                                                          weighted, batch, readout);
}
